// BioGuardGAT_72722386256301
// MI455X (gfx1250) — matrix operand order repaired and checked by reading, not compiled, not run on hardware
//
#include <hip/hip_runtime.h>
#include <hip/hip_bf16.h>

typedef __attribute__((ext_vector_type(16))) _Float16 v16h;
typedef __attribute__((ext_vector_type(8)))  _Float16 v8h;
typedef __attribute__((ext_vector_type(8)))  float    v8f;

// ---------------- helpers ----------------
__device__ __forceinline__ unsigned fenc(float f) {
  unsigned u = __float_as_uint(f);
  return (u & 0x80000000u) ? ~u : (u | 0x80000000u);
}
__device__ __forceinline__ float fdec(unsigned u) {
  u = (u & 0x80000000u) ? (u & 0x7FFFFFFFu) : ~u;
  return __uint_as_float(u);
}

__global__ void fill_f32_kernel(float* p, float v, long long n) {
  long long i = (long long)blockIdx.x * blockDim.x + threadIdx.x;
  if (i < n) p[i] = v;
}
__global__ void fill_u32_kernel(unsigned* p, unsigned v, long long n) {
  long long i = (long long)blockIdx.x * blockDim.x + threadIdx.x;
  if (i < n) p[i] = v;
}
__global__ void f32_to_f16_kernel(const float* __restrict__ s, _Float16* __restrict__ d, long long n) {
  long long i = (long long)blockIdx.x * blockDim.x + threadIdx.x;
  if (i < n) d[i] = (_Float16)s[i];
}
// W [K][Nc] f32 -> Wt [Nc][K] f16 (transpose for contiguous per-column K loads)
__global__ void wt_convert_kernel(const float* __restrict__ W, _Float16* __restrict__ Wt, int K, int Nc) {
  int idx = blockIdx.x * blockDim.x + threadIdx.x;
  if (idx >= K * Nc) return;
  int n = idx / K, k = idx % K;
  Wt[(size_t)n * K + k] = (_Float16)W[(size_t)k * Nc + n];
}

// ---------------- WMMA GEMM: C[M,Nc] = A16[M,K] @ Bt16[Nc,K]^T ----------------
// M % 16 == 0, K % 32 == 0, Nc % 64 == 0.
// 8 waves/block. Each wave owns a 16x64 C tile (4 accumulators): the A fragment
// is loaded once per k-step and reused across 4 WMMAs (4x less A traffic).
// Block stages a 64x32 B tile (4 KB) in LDS, one b128 copy per thread.
__global__ __launch_bounds__(256) void wmma_gemm_kernel(
    const _Float16* __restrict__ A, const _Float16* __restrict__ Bt,
    float* __restrict__ C, int M, int K, int Nc) {
  __shared__ __align__(16) _Float16 ldsB[64 * 32];
  const int tid  = threadIdx.x;
  const int wave = tid >> 5;
  const int lane = tid & 31;
  const int nb0  = blockIdx.x * 64;          // first output column of block
  const int tm   = blockIdx.y * 8 + wave;    // row tile
  const bool active = (tm * 16) < M;

  const int mrow = tm * 16 + (lane & 15);
  const int ka   = (lane >> 4) * 16;         // A: lanes 0-15 K[0..15], lanes 16-31 K[16..31]
  const int nb   = lane & 15;                // B fragment column within sub-tile
  const int kb   = (lane >> 4) * 16;         // B: lanes 0-15 K[0..15], lanes 16-31 K[16..31]

  // staging indices: thread copies one 16-byte chunk of the 64x32 B tile
  const int sn = tid >> 2;                   // 0..63
  const int sk = (tid & 3) * 8;              // 0,8,16,24

  v8f acc0 = {}, acc1 = {}, acc2 = {}, acc3 = {};
  for (int k0 = 0; k0 < K; k0 += 32) {
    *(v8h*)(ldsB + sn * 32 + sk) =
        *(const v8h*)(Bt + (size_t)(nb0 + sn) * K + k0 + sk);
    __syncthreads();
    if (active) {
      const _Float16* ap = A + (size_t)mrow * K + k0 + ka;
      if (k0 + 32 < K) __builtin_prefetch(ap + 32, 0, 0);
      union { v16h v; v8h h[2]; } a, b0, b1, b2, b3;
      a.h[0] = *(const v8h*)(ap);
      a.h[1] = *(const v8h*)(ap + 8);
      const _Float16* bp = ldsB + nb * 32 + kb;
      b0.h[0] = *(const v8h*)(bp);
      b0.h[1] = *(const v8h*)(bp + 8);
      b1.h[0] = *(const v8h*)(bp + 16 * 32);
      b1.h[1] = *(const v8h*)(bp + 16 * 32 + 8);
      b2.h[0] = *(const v8h*)(bp + 32 * 32);
      b2.h[1] = *(const v8h*)(bp + 32 * 32 + 8);
      b3.h[0] = *(const v8h*)(bp + 48 * 32);
      b3.h[1] = *(const v8h*)(bp + 48 * 32 + 8);
      acc0 = __builtin_amdgcn_wmma_f32_16x16x32_f16(false, a.v, false, b0.v, (short)0, acc0, false, false);
      acc1 = __builtin_amdgcn_wmma_f32_16x16x32_f16(false, a.v, false, b1.v, (short)0, acc1, false, false);
      acc2 = __builtin_amdgcn_wmma_f32_16x16x32_f16(false, a.v, false, b2.v, (short)0, acc2, false, false);
      acc3 = __builtin_amdgcn_wmma_f32_16x16x32_f16(false, a.v, false, b3.v, (short)0, acc3, false, false);
    }
    __syncthreads();
  }
  if (active) {
    const int nlocal = lane & 15;
    const int mbase  = (lane >> 4) * 8;
    float* crow = C + (size_t)(tm * 16 + mbase) * Nc + nb0 + nlocal;
#pragma unroll
    for (int r = 0; r < 8; ++r) {
      crow[(size_t)r * Nc]      = acc0[r];
      crow[(size_t)r * Nc + 16] = acc1[r];
      crow[(size_t)r * Nc + 32] = acc2[r];
      crow[(size_t)r * Nc + 48] = acc3[r];
    }
  }
}

// ---------------- elementwise epilogues ----------------
// Y = relu(bn(Y + bias)); optional f16 copy. bias may be null (plain bn+relu).
__global__ void bias_bn_relu_kernel(float* Y, const float* bias, const float* g,
                                    const float* beta, const float* mean, const float* var,
                                    _Float16* y16, long long total, int Nc) {
  long long i = (long long)blockIdx.x * blockDim.x + threadIdx.x;
  if (i >= total) return;
  int n = (int)(i % Nc);
  float v = Y[i];
  if (bias) v += bias[n];
  v = (v - mean[n]) * rsqrtf(var[n] + 1e-5f) * g[n] + beta[n];
  v = fmaxf(v, 0.f);
  Y[i] = v;
  if (y16) y16[i] = (_Float16)v;
}
__global__ void add_bias_kernel(float* Y, const float* b, long long total, int Nc) {
  long long i = (long long)blockIdx.x * blockDim.x + threadIdx.x;
  if (i >= total) return;
  Y[i] += b[(int)(i % Nc)];
}
__global__ void bias_elu_kernel(float* Y, const float* b, _Float16* y16, long long total, int Nc) {
  long long i = (long long)blockIdx.x * blockDim.x + threadIdx.x;
  if (i >= total) return;
  float v = Y[i] + b[(int)(i % Nc)];
  v = v > 0.f ? v : (expf(v) - 1.f);
  Y[i] = v;
  if (y16) y16[i] = (_Float16)v;
}

// ---------------- GATv2 sparse stages ----------------
__global__ void edge_attr_sum_kernel(const int* __restrict__ dst, const float* __restrict__ ea,
                                     float* deg, float* lsum, int E) {
  int idx = blockIdx.x * blockDim.x + threadIdx.x;
  if (idx >= E * 16) return;
  int e = idx >> 4, j = idx & 15;
  int d = dst[e];
  atomicAdd(&lsum[(size_t)d * 16 + j], ea[(size_t)e * 16 + j]);
  if (j == 0) atomicAdd(&deg[d], 1.f);
}
__global__ void loopattr_fin_kernel(float* lsum, const float* deg, int N) {
  int idx = blockIdx.x * blockDim.x + threadIdx.x;
  if (idx >= N * 16) return;
  lsum[idx] /= fmaxf(deg[idx >> 4], 1.f);
}

// one wave per edge i in [0, E+N): logit[i][h] = sum_c lrelu(xl[s]+xr[d]+ee)*att[h][c]
__global__ void gat_logits_kernel(const float* __restrict__ xl, const float* __restrict__ xr,
                                  const int* __restrict__ src, const int* __restrict__ dst,
                                  const float* __restrict__ ea, const float* __restrict__ loopattr,
                                  const float* __restrict__ We, const float* __restrict__ att,
                                  float* __restrict__ logit, unsigned* __restrict__ mxenc,
                                  int E, int N, int H, int C) {
  int i = (blockIdx.x * blockDim.x + threadIdx.x) >> 5;
  int lane = threadIdx.x & 31;
  if (i >= E + N) return;
  int s, d;
  const float* attr = nullptr;
  if (i < E) { s = src[i]; d = dst[i]; attr = ea + (size_t)i * 16; }
  else       { s = i - E; d = s; if (loopattr) attr = loopattr + (size_t)s * 16; }
  int HC = H * C;
  for (int h = 0; h < H; ++h) {
    float part = 0.f;
    for (int c = lane; c < C; c += 32) {
      float ee = 0.f;
      if (attr) {
#pragma unroll
        for (int j = 0; j < 16; ++j) ee += attr[j] * We[(size_t)j * HC + h * C + c];
      }
      float m = xl[(size_t)s * HC + h * C + c] + xr[(size_t)d * HC + h * C + c] + ee;
      m = m > 0.f ? m : 0.2f * m;
      part += m * att[h * C + c];
    }
    for (int o = 16; o > 0; o >>= 1) part += __shfl_down(part, o);
    if (lane == 0) {
      logit[(size_t)i * H + h] = part;
      atomicMax(&mxenc[(size_t)d * H + h], fenc(part));
    }
  }
}

__global__ void gat_exp_kernel(const float* __restrict__ logit, const unsigned* __restrict__ mxenc,
                               const int* __restrict__ dst, float* __restrict__ aexp,
                               float* __restrict__ ssum, int E, int N, int H) {
  long long idx = (long long)blockIdx.x * blockDim.x + threadIdx.x;
  if (idx >= (long long)(E + N) * H) return;
  int i = (int)(idx / H), h = (int)(idx % H);
  int d = (i < E) ? dst[i] : (i - E);
  float a = expf(logit[idx] - fdec(mxenc[(size_t)d * H + h]));
  aexp[idx] = a;
  atomicAdd(&ssum[(size_t)d * H + h], a);
}

// one wave per edge: out[d] += (aexp/ssum[d]) * xl[s]
__global__ void gat_accum_kernel(const float* __restrict__ xl, const float* __restrict__ aexp,
                                 const float* __restrict__ ssum, const int* __restrict__ src,
                                 const int* __restrict__ dst, float* __restrict__ out,
                                 int E, int N, int H, int C) {
  int i = (blockIdx.x * blockDim.x + threadIdx.x) >> 5;
  int lane = threadIdx.x & 31;
  if (i >= E + N) return;
  int s, d;
  if (i < E) { s = src[i]; d = dst[i]; } else { s = i - E; d = s; }
  int HC = H * C;
  float w[4];
  for (int h = 0; h < H; ++h)
    w[h] = aexp[(size_t)i * H + h] / fmaxf(ssum[(size_t)d * H + h], 1e-20f);
  for (int idx = lane; idx < HC; idx += 32) {
    int h = idx / C;
    atomicAdd(&out[(size_t)d * HC + idx], w[h] * xl[(size_t)s * HC + idx]);
  }
}

// ---------------- pooling ----------------
__global__ void pool_kernel(const float* __restrict__ h, const int* __restrict__ batch,
                            float* cnt, float* meanacc, unsigned* maxenc, int N, int C) {
  long long idx = (long long)blockIdx.x * blockDim.x + threadIdx.x;
  if (idx >= (long long)N * C) return;
  int i = (int)(idx / C), c = (int)(idx % C);
  int g = batch[i];
  float v = h[idx];
  atomicAdd(&meanacc[(size_t)g * C + c], v);
  atomicMax(&maxenc[(size_t)g * C + c], fenc(v));
  if (c == 0) atomicAdd(&cnt[g], 1.f);
}
__global__ void pool_fin_kernel(const float* cnt, const float* meanacc, const unsigned* maxenc,
                                float* v, int G, int C) {
  int idx = blockIdx.x * blockDim.x + threadIdx.x;
  if (idx >= G * C) return;
  int g = idx / C, c = idx % C;
  float ct = cnt[g];
  v[(size_t)g * 2 * C + c]     = meanacc[idx] / fmaxf(ct, 1.f);
  v[(size_t)g * 2 * C + C + c] = (ct > 0.f) ? fdec(maxenc[idx]) : 0.f;
}

// ---------------- head ----------------
// gate_in[g] = [sum(256), abs(256), prod(256), esum(15), eabs(15), eprod(15)] (width 813)
__global__ void build_feats_kernel(const float* va, const float* vb,
                                   const float* enza, const float* enzb,
                                   float* gate_in, int G) {
  int idx = blockIdx.x * blockDim.x + threadIdx.x;
  if (idx >= G * 813) return;
  int g = idx / 813, j = idx % 813;
  float A, B;
  int part;
  if (j < 768) { int k = j % 256; part = j / 256; A = va[(size_t)g * 256 + k]; B = vb[(size_t)g * 256 + k]; }
  else { int jj = j - 768; int k = jj % 15; part = jj / 15; A = enza[(size_t)g * 15 + k]; B = enzb[(size_t)g * 15 + k]; }
  float r = (part == 0) ? (A + B) : (part == 1) ? fabsf(A - B) : (A * B);
  gate_in[(size_t)g * 813 + j] = r;
}

__global__ void dense_kernel(const float* __restrict__ X, int ldx, const float* __restrict__ W,
                             const float* __restrict__ b, float* __restrict__ Y,
                             int M, int K, int Nc, int act) {
  int idx = blockIdx.x * blockDim.x + threadIdx.x;
  if (idx >= M * Nc) return;
  int m = idx / Nc, n = idx % Nc;
  float acc = b ? b[n] : 0.f;
  for (int k = 0; k < K; ++k) acc += X[(size_t)m * ldx + k] * W[(size_t)k * Nc + n];
  if (act == 1) acc = fmaxf(acc, 0.f);
  else if (act == 2) acc = 1.f / (1.f + expf(-acc));
  Y[(size_t)m * Nc + n] = acc;
}

__global__ void final_combine_kernel(const float* alpha, const float* gl, const float* pr,
                                     float* out, int G) {
  int g = blockIdx.x * blockDim.x + threadIdx.x;
  if (g >= G) return;
  float a = alpha[g];
  out[g]     = a * gl[g] + (1.f - a) * pr[g];
  out[G + g] = a;
}

// ---------------- host ----------------
static inline unsigned gx(long long n) { return (unsigned)((n + 255) / 256); }

extern "C" void kernel_launch(void* const* d_in, const int* in_sizes, int n_in,
                              void* d_out, int out_size, void* d_ws, size_t ws_size,
                              hipStream_t stream) {
  (void)n_in; (void)ws_size; (void)out_size;
  const int N = in_sizes[0] / 64;
  const int E = in_sizes[1] / 2;
  const int G = in_sizes[4] / 15;

  const float* xA   = (const float*)d_in[0];
  const int*   eiA  = (const int*)  d_in[1];
  const float* eaA  = (const float*)d_in[2];
  const int*   bA   = (const int*)  d_in[3];
  const float* enzA = (const float*)d_in[4];
  const float* xB   = (const float*)d_in[5];
  const int*   eiB  = (const int*)  d_in[6];
  const float* eaB  = (const float*)d_in[7];
  const int*   bB   = (const int*)  d_in[8];
  const float* enzB = (const float*)d_in[9];
  const float* aW = (const float*)d_in[10], *ab = (const float*)d_in[11];
  const float* ag = (const float*)d_in[12], *abeta = (const float*)d_in[13];
  const float* am = (const float*)d_in[14], *av = (const float*)d_in[15];
  const float* Wl1 = (const float*)d_in[16], *bl1 = (const float*)d_in[17];
  const float* Wr1 = (const float*)d_in[18], *br1 = (const float*)d_in[19];
  const float* We1 = (const float*)d_in[20], *att1 = (const float*)d_in[21];
  const float* bias1 = (const float*)d_in[22];
  const float* Wl2 = (const float*)d_in[23], *bl2 = (const float*)d_in[24];
  const float* Wr2 = (const float*)d_in[25], *br2 = (const float*)d_in[26];
  const float* We2 = (const float*)d_in[27], *att2 = (const float*)d_in[28];
  const float* bias2 = (const float*)d_in[29];
  const float* gW1 = (const float*)d_in[30], *gb1 = (const float*)d_in[31];
  const float* gg = (const float*)d_in[32], *gbeta = (const float*)d_in[33];
  const float* gm = (const float*)d_in[34], *gv = (const float*)d_in[35];
  const float* gW2 = (const float*)d_in[36], *gb2 = (const float*)d_in[37];
  const float* gW3 = (const float*)d_in[38], *gb3 = (const float*)d_in[39];
  const float* pW1 = (const float*)d_in[40], *pb1 = (const float*)d_in[41];
  const float* pW2 = (const float*)d_in[42], *pb2 = (const float*)d_in[43];
  const float* qW1 = (const float*)d_in[44], *qb1 = (const float*)d_in[45];
  const float* qW2 = (const float*)d_in[46], *qb2 = (const float*)d_in[47];

  // bump allocator on workspace
  char* ws = (char*)d_ws;
  size_t off = 0;
  auto alloc = [&](size_t bytes) -> void* {
    off = (off + 255) & ~(size_t)255;
    void* p = ws + off;
    off += bytes;
    return p;
  };

  // persistent
  float* va = (float*)alloc((size_t)G * 256 * 4);
  float* vb = (float*)alloc((size_t)G * 256 * 4);
  _Float16* aWt  = (_Float16*)alloc((size_t)64 * 128 * 2);
  _Float16* Wl1t = (_Float16*)alloc((size_t)128 * 512 * 2);
  _Float16* Wr1t = (_Float16*)alloc((size_t)128 * 512 * 2);
  _Float16* Wl2t = (_Float16*)alloc((size_t)512 * 128 * 2);
  _Float16* Wr2t = (_Float16*)alloc((size_t)512 * 128 * 2);
  // per-arm scratch (reused for both arms)
  _Float16* x16  = (_Float16*)alloc((size_t)N * 64 * 2);
  _Float16* h16  = (_Float16*)alloc((size_t)N * 512 * 2);
  float* bufH0 = (float*)alloc((size_t)N * 128 * 4);
  float* bufA  = (float*)alloc((size_t)N * 512 * 4);
  float* bufB  = (float*)alloc((size_t)N * 512 * 4);
  float* bufC  = (float*)alloc((size_t)N * 512 * 4);
  float* logit = (float*)alloc((size_t)(E + N) * 4 * 4);
  float* aexp  = (float*)alloc((size_t)(E + N) * 4 * 4);
  unsigned* mxenc = (unsigned*)alloc((size_t)N * 4 * 4);
  float* ssum  = (float*)alloc((size_t)N * 4 * 4);
  float* deg   = (float*)alloc((size_t)N * 4);
  float* lsum  = (float*)alloc((size_t)N * 16 * 4);
  float* cnt   = (float*)alloc((size_t)G * 4);
  float* meanacc = (float*)alloc((size_t)G * 128 * 4);
  unsigned* maxg = (unsigned*)alloc((size_t)G * 128 * 4);
  // head
  float* gate_in = (float*)alloc((size_t)G * 813 * 4);
  float* hh1 = (float*)alloc((size_t)G * 256 * 4);
  float* hh2 = (float*)alloc((size_t)G * 64 * 4);
  float* glog = (float*)alloc((size_t)G * 4);
  float* ph = (float*)alloc((size_t)G * 64 * 4);
  float* prior = (float*)alloc((size_t)G * 4);
  float* qh = (float*)alloc((size_t)G * 32 * 4);
  float* alpha = (float*)alloc((size_t)G * 4);

  // weight conversions (transpose to [Nc][K] f16)
  wt_convert_kernel<<<gx(64 * 128), 256, 0, stream>>>(aW, aWt, 64, 128);
  wt_convert_kernel<<<gx(128 * 512), 256, 0, stream>>>(Wl1, Wl1t, 128, 512);
  wt_convert_kernel<<<gx(128 * 512), 256, 0, stream>>>(Wr1, Wr1t, 128, 512);
  wt_convert_kernel<<<gx(512 * 128), 256, 0, stream>>>(Wl2, Wl2t, 512, 128);
  wt_convert_kernel<<<gx(512 * 128), 256, 0, stream>>>(Wr2, Wr2t, 512, 128);

  auto gemm = [&](const _Float16* A16, const _Float16* Bt, float* C, int M, int K, int Nc) {
    dim3 grid(Nc / 64, (unsigned)((M / 16 + 7) / 8));
    wmma_gemm_kernel<<<grid, 256, 0, stream>>>(A16, Bt, C, M, K, Nc);
  };

  auto run_arm = [&](const float* x, const int* ei, const float* ea, const int* batch, float* vout) {
    const int* src = ei;
    const int* dst = ei + E;
    long long EN = (long long)E + N;
    // node encoder: h0 = relu(bn(x@aW + ab))
    f32_to_f16_kernel<<<gx((long long)N * 64), 256, 0, stream>>>(x, x16, (long long)N * 64);
    gemm(x16, aWt, bufH0, N, 64, 128);
    bias_bn_relu_kernel<<<gx((long long)N * 128), 256, 0, stream>>>(
        bufH0, ab, ag, abeta, am, av, h16, (long long)N * 128, 128);
    // loop_attr = segmean of incoming edge_attr
    fill_f32_kernel<<<gx(N), 256, 0, stream>>>(deg, 0.f, N);
    fill_f32_kernel<<<gx((long long)N * 16), 256, 0, stream>>>(lsum, 0.f, (long long)N * 16);
    edge_attr_sum_kernel<<<gx((long long)E * 16), 256, 0, stream>>>(dst, ea, deg, lsum, E);
    loopattr_fin_kernel<<<gx((long long)N * 16), 256, 0, stream>>>(lsum, deg, N);
    // conv1 (H=4, C=128)
    gemm(h16, Wl1t, bufA, N, 128, 512);
    add_bias_kernel<<<gx((long long)N * 512), 256, 0, stream>>>(bufA, bl1, (long long)N * 512, 512);
    gemm(h16, Wr1t, bufB, N, 128, 512);
    add_bias_kernel<<<gx((long long)N * 512), 256, 0, stream>>>(bufB, br1, (long long)N * 512, 512);
    fill_u32_kernel<<<gx((long long)N * 4), 256, 0, stream>>>(mxenc, 0u, (long long)N * 4);
    gat_logits_kernel<<<gx(EN * 32), 256, 0, stream>>>(bufA, bufB, src, dst, ea, lsum,
                                                       We1, att1, logit, mxenc, E, N, 4, 128);
    fill_f32_kernel<<<gx((long long)N * 4), 256, 0, stream>>>(ssum, 0.f, (long long)N * 4);
    gat_exp_kernel<<<gx(EN * 4), 256, 0, stream>>>(logit, mxenc, dst, aexp, ssum, E, N, 4);
    fill_f32_kernel<<<gx((long long)N * 512), 256, 0, stream>>>(bufC, 0.f, (long long)N * 512);
    gat_accum_kernel<<<gx(EN * 32), 256, 0, stream>>>(bufA, aexp, ssum, src, dst, bufC, E, N, 4, 128);
    bias_elu_kernel<<<gx((long long)N * 512), 256, 0, stream>>>(bufC, bias1, h16, (long long)N * 512, 512);
    // conv2 (H=1, C=128), no fill_mean (zero self-loop attr)
    gemm(h16, Wl2t, bufA, N, 512, 128);
    add_bias_kernel<<<gx((long long)N * 128), 256, 0, stream>>>(bufA, bl2, (long long)N * 128, 128);
    gemm(h16, Wr2t, bufB, N, 512, 128);
    add_bias_kernel<<<gx((long long)N * 128), 256, 0, stream>>>(bufB, br2, (long long)N * 128, 128);
    fill_u32_kernel<<<gx(N), 256, 0, stream>>>(mxenc, 0u, N);
    gat_logits_kernel<<<gx(EN * 32), 256, 0, stream>>>(bufA, bufB, src, dst, ea, nullptr,
                                                       We2, att2, logit, mxenc, E, N, 1, 128);
    fill_f32_kernel<<<gx(N), 256, 0, stream>>>(ssum, 0.f, N);
    gat_exp_kernel<<<gx(EN), 256, 0, stream>>>(logit, mxenc, dst, aexp, ssum, E, N, 1);
    fill_f32_kernel<<<gx((long long)N * 128), 256, 0, stream>>>(bufC, 0.f, (long long)N * 128);
    gat_accum_kernel<<<gx(EN * 32), 256, 0, stream>>>(bufA, aexp, ssum, src, dst, bufC, E, N, 1, 128);
    bias_elu_kernel<<<gx((long long)N * 128), 256, 0, stream>>>(bufC, bias2, nullptr, (long long)N * 128, 128);
    // pooling -> vout [G, 256]
    fill_f32_kernel<<<gx(G), 256, 0, stream>>>(cnt, 0.f, G);
    fill_f32_kernel<<<gx((long long)G * 128), 256, 0, stream>>>(meanacc, 0.f, (long long)G * 128);
    fill_u32_kernel<<<gx((long long)G * 128), 256, 0, stream>>>(maxg, 0u, (long long)G * 128);
    pool_kernel<<<gx((long long)N * 128), 256, 0, stream>>>(bufC, batch, cnt, meanacc, maxg, N, 128);
    pool_fin_kernel<<<gx((long long)G * 128), 256, 0, stream>>>(cnt, meanacc, maxg, vout, G, 128);
  };

  run_arm(xA, eiA, eaA, bA, va);
  run_arm(xB, eiB, eaB, bB, vb);

  // head
  build_feats_kernel<<<gx((long long)G * 813), 256, 0, stream>>>(va, vb, enzA, enzB, gate_in, G);
  dense_kernel<<<gx(G * 256), 256, 0, stream>>>(gate_in, 813, gW1, gb1, hh1, G, 768, 256, 0);
  bias_bn_relu_kernel<<<gx((long long)G * 256), 256, 0, stream>>>(
      hh1, nullptr, gg, gbeta, gm, gv, nullptr, (long long)G * 256, 256);
  dense_kernel<<<gx(G * 64), 256, 0, stream>>>(hh1, 256, gW2, gb2, hh2, G, 256, 64, 1);
  dense_kernel<<<gx(G), 256, 0, stream>>>(hh2, 64, gW3, gb3, glog, G, 64, 1, 0);
  dense_kernel<<<gx(G * 64), 256, 0, stream>>>(gate_in + 768, 813, pW1, pb1, ph, G, 45, 64, 1);
  dense_kernel<<<gx(G), 256, 0, stream>>>(ph, 64, pW2, pb2, prior, G, 64, 1, 0);
  dense_kernel<<<gx(G * 32), 256, 0, stream>>>(gate_in, 813, qW1, qb1, qh, G, 813, 32, 1);
  dense_kernel<<<gx(G), 256, 0, stream>>>(qh, 32, qW2, qb2, alpha, G, 32, 1, 2);
  final_combine_kernel<<<gx(G), 256, 0, stream>>>(alpha, glog, prior, (float*)d_out, G);
}
